// MultiHeadAttention_71562745085965
// MI455X (gfx1250) — matrix operand order repaired and checked by reading, not compiled, not run on hardware
//
#include <hip/hip_runtime.h>

typedef __attribute__((ext_vector_type(16))) _Float16 v16h;
typedef __attribute__((ext_vector_type(8)))  _Float16 v8h;
typedef __attribute__((ext_vector_type(4)))  _Float16 v4h;
typedef __attribute__((ext_vector_type(8)))  float    v8f;

#define BB 2
#define LL 2048
#define DD 1024
#define HH 16
#define DHH 64

__device__ __forceinline__ v8f wmma_f16(v16h a, v16h b, v8f c) {
    return __builtin_amdgcn_wmma_f32_16x16x32_f16(false, a, false, b, (short)0, c,
                                                  false, false);
}

// A-matrix 16x32 f16 fragment from row-major block, row stride ld (halves).
__device__ __forceinline__ v16h load_a_frag(const _Float16* base, int ld, int lane) {
    int nlo = lane & 15, hi = (lane >> 4) & 1;
    const _Float16* p = base + (size_t)nlo * ld + hi * 16;
    v8h lo = *(const v8h*)(p);
    v8h hh = *(const v8h*)(p + 8);
    v16h a;
#pragma unroll
    for (int i = 0; i < 8; ++i) { a[i] = lo[i]; a[i + 8] = hh[i]; }
    return a;
}

// B-matrix 32x16 fragment of (row-major M)^T: contiguous 32B per lane.
__device__ __forceinline__ v16h load_bt_frag(const _Float16* base, int ld, int lane) {
    int nlo = lane & 15, hi = (lane >> 4) & 1;
    return *(const v16h*)(base + (size_t)nlo * ld + hi * 16);
}

// Async 16B global->LDS copy (global_load_async_to_lds_b128, ASYNCcnt-tracked).
__device__ __forceinline__ void async_copy16(_Float16* lds_dst, const _Float16* src) {
    unsigned dst = (unsigned)(unsigned long long)lds_dst;
    asm volatile("global_load_async_to_lds_b128 %0, %1, off"
                 :: "v"(dst), "v"(src) : "memory");
}
__device__ __forceinline__ void wait_async() {
    asm volatile("s_wait_asynccnt 0" ::: "memory");
}

// All-reduce max within each 16-lane row via DPP16 row_ror (no LDS traffic).
__device__ __forceinline__ float rowmax16(float x) {
    int y;
    y = __builtin_amdgcn_update_dpp(0, __builtin_bit_cast(int, x), 0x121, 0xf, 0xf, true);
    x = fmaxf(x, __builtin_bit_cast(float, y));
    y = __builtin_amdgcn_update_dpp(0, __builtin_bit_cast(int, x), 0x122, 0xf, 0xf, true);
    x = fmaxf(x, __builtin_bit_cast(float, y));
    y = __builtin_amdgcn_update_dpp(0, __builtin_bit_cast(int, x), 0x124, 0xf, 0xf, true);
    x = fmaxf(x, __builtin_bit_cast(float, y));
    y = __builtin_amdgcn_update_dpp(0, __builtin_bit_cast(int, x), 0x128, 0xf, 0xf, true);
    x = fmaxf(x, __builtin_bit_cast(float, y));
    return x;
}

// ---------------------------------------------------------------- f32 -> f16
__global__ __launch_bounds__(256) void cvt_f32_f16(const float4* __restrict__ s,
                                                   v4h* __restrict__ d, int n4) {
    int i = blockIdx.x * blockDim.x + threadIdx.x;
    if (i < n4) {
        float4 f = s[i];
        v4h h = {(_Float16)f.x, (_Float16)f.y, (_Float16)f.z, (_Float16)f.w};
        d[i] = h;
    }
}

// ------------------------------------------------- Y = X[M,K] * W[N,K]^T
// One wave computes a 64x64 tile (4 A frags x 4 B frags -> 16 WMMA per k-step).
// MODE: 0 = f16 [M,N]; 1 = f16 transposed per batch [B][N][L]; 2 = f32 [M,N].
template <int MODE>
__global__ __launch_bounds__(128) void gemm_xwT(const _Float16* __restrict__ X,
                                                const _Float16* __restrict__ W,
                                                void* __restrict__ Y,
                                                int M, int N, int K) {
    int wave = (blockIdx.x * blockDim.x + threadIdx.x) >> 5;
    int lane = threadIdx.x & 31;
    int mt = M >> 6;                 // 64-row tiles
    int tm = wave % mt;
    int tn = wave / mt;
    if (tn >= (N >> 6)) return;

    const _Float16* Xb = X + (size_t)tm * 64 * K;
    v8f acc[4][4] = {};              // [mi][c]
    for (int k0 = 0; k0 < K; k0 += 32) {
        v16h a[4];
#pragma unroll
        for (int mi = 0; mi < 4; ++mi)
            a[mi] = load_a_frag(Xb + (size_t)mi * 16 * K + k0, K, lane);
#pragma unroll
        for (int c = 0; c < 4; ++c) {
            v16h b = load_bt_frag(W + (size_t)(tn * 64 + c * 16) * K + k0, K, lane);
#pragma unroll
            for (int mi = 0; mi < 4; ++mi)
                acc[mi][c] = wmma_f16(a[mi], b, acc[mi][c]);
        }
    }
    int nlo = lane & 15, hi = (lane >> 4) & 1;
#pragma unroll
    for (int mi = 0; mi < 4; ++mi)
#pragma unroll
        for (int c = 0; c < 4; ++c)
#pragma unroll
            for (int r = 0; r < 8; ++r) {
                float v = acc[mi][c][r];
                size_t m = (size_t)tm * 64 + mi * 16 + r + hi * 8;
                size_t n = (size_t)tn * 64 + c * 16 + nlo;
                if (MODE == 0) ((_Float16*)Y)[m * N + n] = (_Float16)v;
                else if (MODE == 1) {   // V^T per batch: [b][n][l]
                    size_t b = m >> 11, l = m & (LL - 1);
                    ((_Float16*)Y)[(b * DD + n) * LL + l] = (_Float16)v;
                } else ((float*)Y)[m * N + n] = v;
            }
}

// --------------------------------------------------------- causal flash attn
// Q,K,O: [B*L, D] f16; VT: [B][D][L] f16. Block = 4 waves = 64 q rows of one
// (b,h). 32-key K/V^T tiles staged in LDS via async copies, double-buffered.
__global__ __launch_bounds__(128) void flash_attn(const _Float16* __restrict__ Q,
                                                  const _Float16* __restrict__ Kh,
                                                  const _Float16* __restrict__ VT,
                                                  _Float16* __restrict__ O) {
    __shared__ __align__(32) _Float16 Kbuf[2][32][64];  // [key][dh]
    __shared__ __align__(32) _Float16 Vbuf[2][64][32];  // [dh][key]
    __shared__ __align__(32) _Float16 Pt[4][16][32];

    int t    = threadIdx.x;
    int w    = t >> 5;
    int lane = t & 31;
    int nlo = lane & 15, hi = (lane >> 4) & 1;

    int blocksPerSeq = LL / 64;
    int q0b = (blockIdx.x % blocksPerSeq) * 64;
    int bh  = blockIdx.x / blocksPerSeq;
    int h = bh % HH, b = bh / HH;
    size_t rowbase = (size_t)b * LL;
    int colbase = h * DHH;
    int q0 = q0b + w * 16;

    const _Float16* Qb = Q + (rowbase + q0) * DD + colbase;
    v16h aq0 = load_a_frag(Qb, DD, lane);
    v16h aq1 = load_a_frag(Qb + 32, DD, lane);
#pragma unroll
    for (int e = 0; e < 16; ++e) {          // fold 1/sqrt(64) into Q (exact)
        aq0[e] = aq0[e] * (_Float16)0.125f;
        aq1[e] = aq1[e] * (_Float16)0.125f;
    }
    v16h ones;
#pragma unroll
    for (int e = 0; e < 16; ++e) ones[e] = (_Float16)1.0f;

    float mrow[8], lrow[8], scl[8];
    v8f o[4] = {};
#pragma unroll
    for (int r = 0; r < 8; ++r) { mrow[r] = -3.0e38f; lrow[r] = 0.f; }

    const int qend  = q0 + 15;
    const int ktmax = q0b + 63;

    auto stage = [&](int kt, int p) {
        for (int s = t; s < 256; s += 128) {           // K tile: 256 x 16B
            int r = s >> 3, off = (s & 7) * 8;
            async_copy16(&Kbuf[p][r][off],
                         Kh + (rowbase + kt + r) * DD + colbase + off);
        }
        for (int s = t; s < 256; s += 128) {           // VT tile: 256 x 16B
            int j = s >> 2, off = (s & 3) * 8;
            async_copy16(&Vbuf[p][j][off],
                         VT + ((size_t)b * DD + colbase + j) * LL + kt + off);
        }
    };

    // Softmax body; domask is compile-time constant at each call site.
    auto softmax_tile = [&](v8f s0, v8f s1, int kt, bool domask) {
#pragma unroll
        for (int r = 0; r < 8; ++r) {
            float v0 = s0[r];
            float v1 = s1[r];
            if (domask) {
                int rowg = q0 + r + hi * 8;
                if (kt + nlo      > rowg) v0 = -3.0e38f;
                if (kt + 16 + nlo > rowg) v1 = -3.0e38f;
            }
            float mnew = fmaxf(mrow[r], rowmax16(fmaxf(v0, v1)));
            float p0 = __expf(v0 - mnew);
            float p1 = __expf(v1 - mnew);
            scl[r]  = __expf(mrow[r] - mnew);
            mrow[r] = mnew;
            int m = r + hi * 8;
            Pt[w][m][nlo]      = (_Float16)p0;
            Pt[w][m][16 + nlo] = (_Float16)p1;
        }
    };

    stage(0, 0);
    int p = 0;
    for (int kt = 0; kt <= ktmax; kt += 32, p ^= 1) {
        wait_async();
        __syncthreads();                       // buffer p ready for all waves
        if (kt + 32 <= ktmax) stage(kt + 32, p ^ 1);

        if (kt <= qend) {
            v8f s0 = {}, s1 = {};
            s0 = wmma_f16(aq0, load_bt_frag(&Kbuf[p][0][0],  64, lane), s0);
            s0 = wmma_f16(aq1, load_bt_frag(&Kbuf[p][0][32], 64, lane), s0);
            s1 = wmma_f16(aq0, load_bt_frag(&Kbuf[p][16][0], 64, lane), s1);
            s1 = wmma_f16(aq1, load_bt_frag(&Kbuf[p][16][32],64, lane), s1);

            if (kt + 31 > q0) softmax_tile(s0, s1, kt, true);   // diagonal
            else              softmax_tile(s0, s1, kt, false);  // interior

            asm volatile("s_wait_dscnt 0" ::: "memory");   // same-wave LDS RAW
            v16h pa = load_a_frag(&Pt[w][0][0], 32, lane);

            // row sums via P @ ones (replicated into every column)
            v8f rsf = {};
            rsf = wmma_f16(pa, ones, rsf);
#pragma unroll
            for (int r = 0; r < 8; ++r)
                lrow[r] = lrow[r] * scl[r] + rsf[r];

#pragma unroll
            for (int c = 0; c < 4; ++c)
#pragma unroll
                for (int r = 0; r < 8; ++r) o[c][r] *= scl[r];

#pragma unroll
            for (int c = 0; c < 4; ++c)
                o[c] = wmma_f16(pa, load_bt_frag(&Vbuf[p][c * 16][0], 32, lane), o[c]);
        }
        __syncthreads();                       // all waves done with buffer p
    }

    _Float16* Ob = O + (rowbase + q0) * DD + colbase;
#pragma unroll
    for (int r = 0; r < 8; ++r) scl[r] = 1.0f / lrow[r];
#pragma unroll
    for (int c = 0; c < 4; ++c)
#pragma unroll
        for (int r = 0; r < 8; ++r)
            Ob[(size_t)(r + hi * 8) * DD + c * 16 + nlo] = (_Float16)(o[c][r] * scl[r]);
}

extern "C" void kernel_launch(void* const* d_in, const int* in_sizes, int n_in,
                              void* d_out, int out_size, void* d_ws, size_t ws_size,
                              hipStream_t stream) {
    const float* x  = (const float*)d_in[0];
    const float* Wk = (const float*)d_in[1];   // dict order: x, Wk, Wq, Wv, Wo
    const float* Wq = (const float*)d_in[2];
    const float* Wv = (const float*)d_in[3];
    const float* Wo = (const float*)d_in[4];
    float* out = (float*)d_out;

    const size_t nx = (size_t)BB * LL * DD;
    const size_t nw = (size_t)DD * DD;

    char* p = (char*)d_ws;
    _Float16* xh  = (_Float16*)p; p += nx * 2;
    _Float16* wkh = (_Float16*)p; p += nw * 2;
    _Float16* wqh = (_Float16*)p; p += nw * 2;
    _Float16* wvh = (_Float16*)p; p += nw * 2;
    _Float16* woh = (_Float16*)p; p += nw * 2;
    _Float16* qh  = (_Float16*)p; p += nx * 2;
    _Float16* kh  = (_Float16*)p; p += nx * 2;
    _Float16* vt  = (_Float16*)p; p += nx * 2;   // V^T per batch [B][D][L]
    _Float16* ah  = (_Float16*)p; p += nx * 2;

    const int ct = 256;
    cvt_f32_f16<<<(int)(nx / 4 / ct), ct, 0, stream>>>((const float4*)x,  (v4h*)xh,  (int)(nx / 4));
    cvt_f32_f16<<<(int)(nw / 4 / ct), ct, 0, stream>>>((const float4*)Wk, (v4h*)wkh, (int)(nw / 4));
    cvt_f32_f16<<<(int)(nw / 4 / ct), ct, 0, stream>>>((const float4*)Wq, (v4h*)wqh, (int)(nw / 4));
    cvt_f32_f16<<<(int)(nw / 4 / ct), ct, 0, stream>>>((const float4*)Wv, (v4h*)wvh, (int)(nw / 4));
    cvt_f32_f16<<<(int)(nw / 4 / ct), ct, 0, stream>>>((const float4*)Wo, (v4h*)woh, (int)(nw / 4));

    const int M = BB * LL, N = DD, K = DD;
    const int waves = (M / 64) * (N / 64);    // 1024
    dim3 gb(waves / 4), tb(128);
    gemm_xwT<0><<<gb, tb, 0, stream>>>(xh, wqh, qh, M, N, K);
    gemm_xwT<0><<<gb, tb, 0, stream>>>(xh, wkh, kh, M, N, K);
    gemm_xwT<1><<<gb, tb, 0, stream>>>(xh, wvh, vt, M, N, K);

    flash_attn<<<BB * HH * (LL / 64), 128, 0, stream>>>(qh, kh, vt, ah);

    gemm_xwT<2><<<gb, tb, 0, stream>>>(ah, woh, out, M, N, K);
}
